// Block_28372553958032
// MI455X (gfx1250) — compile-verified, matrix operand order checked by reading, not run on hardware
//
#include <hip/hip_runtime.h>
#include <math.h>

// ---------------------------------------------------------------------------
// Types for CDNA5 WMMA (wave32, 16x16x32 bf16 -> f32)
// ---------------------------------------------------------------------------
typedef __bf16 bf16_t;
typedef __attribute__((ext_vector_type(16))) __bf16 v16bf;
typedef __attribute__((ext_vector_type(8)))  float  v8f;

#define N_EMBD  768
#define N_HEADS 12
#define HEAD_D  64
#define BATCH   8
#define SEQ_T   1024
#define N_ENC_  257
#define LN_EPS  1e-5f

// Async global->LDS staging (CDNA5 GLOBAL_LOAD_ASYNC_TO_LDS_*, ASYNCcnt).
#define USE_ASYNC_LDS 1

__device__ __forceinline__ void async_load_b128(unsigned lds_off, const void* gaddr) {
    asm volatile("global_load_async_to_lds_b128 %0, %1, off"
                 :: "v"(lds_off), "v"(gaddr) : "memory");
}
__device__ __forceinline__ void wait_asynccnt0() {
    asm volatile("s_wait_asynccnt 0x0" ::: "memory");
}

// ---------------------------------------------------------------------------
// Fragment loader: 16-bit A/B matrix fragment per CDNA5 ISA layout.
// A (16x32, MxK): lane holds row m = lane&15; kbase = (lane<16)?0:8.
//   element i -> K = kbase + i + (i>=8 ? 8 : 0); K pairs are contiguous,
//   so each of the 8 VGPRs is one aligned 32-bit load (compiler merges
//   them into ds_load_b128 pairs).
// ---------------------------------------------------------------------------
__device__ __forceinline__ v16bf load_frag(const bf16_t* rowptr, int kb) {
    union FragU { v16bf v; unsigned u[8]; } ua;
    const unsigned* p = (const unsigned*)rowptr;
    int o = kb >> 1;              // pair offset: kb=0 -> 0, kb=8 -> 4
#pragma unroll
    for (int j = 0; j < 4; ++j) ua.u[j] = p[o + j];          // K = kb+2j, kb+2j+1
#pragma unroll
    for (int j = 0; j < 4; ++j) ua.u[4 + j] = p[o + 8 + j];  // K = kb+16+2j, +1
    return ua.v;
}

#define WMMA_BF16(A_, B_, C_) \
    __builtin_amdgcn_wmma_f32_16x16x32_bf16(false, (A_), false, (B_), (short)0, (C_), false, false)

// ---------------------------------------------------------------------------
// f32 -> bf16 elementwise conversion
// ---------------------------------------------------------------------------
__global__ void f32_to_bf16_kernel(const float* __restrict__ in,
                                   bf16_t* __restrict__ out, int n) {
    int i = blockIdx.x * blockDim.x + threadIdx.x;
    if (i < n) out[i] = (bf16_t)in[i];
}

// ---------------------------------------------------------------------------
// LayerNorm: one 256-thread block per row of 768, f32 in -> bf16 out
// ---------------------------------------------------------------------------
__global__ void layernorm_kernel(const float* __restrict__ x,
                                 const float* __restrict__ g,
                                 const float* __restrict__ bb,
                                 bf16_t* __restrict__ out, int rows) {
    __shared__ float red[256];
    int row = blockIdx.x;
    if (row >= rows) return;
    const float* xr = x + (size_t)row * N_EMBD;
    int t = threadIdx.x;
    float v0 = xr[t], v1 = xr[t + 256], v2 = xr[t + 512];

    red[t] = v0 + v1 + v2;
    __syncthreads();
    for (int st = 128; st > 0; st >>= 1) {
        if (t < st) red[t] += red[t + st];
        __syncthreads();
    }
    float mean = red[0] * (1.0f / N_EMBD);
    __syncthreads();

    float d0 = v0 - mean, d1 = v1 - mean, d2 = v2 - mean;
    red[t] = d0 * d0 + d1 * d1 + d2 * d2;
    __syncthreads();
    for (int st = 128; st > 0; st >>= 1) {
        if (t < st) red[t] += red[t + st];
        __syncthreads();
    }
    float rstd = rsqrtf(red[0] * (1.0f / N_EMBD) + LN_EPS);

    bf16_t* orow = out + (size_t)row * N_EMBD;
    orow[t]       = (bf16_t)(d0 * rstd * g[t]       + bb[t]);
    orow[t + 256] = (bf16_t)(d1 * rstd * g[t + 256] + bb[t + 256]);
    orow[t + 512] = (bf16_t)(d2 * rstd * g[t + 512] + bb[t + 512]);
}

// ---------------------------------------------------------------------------
// WMMA GEMM:  C[M,N] = act(A[M,K] @ W[K,N] + bias)  (bf16 A/W, f32 accum)
//   RESID=true:  Cf[M,N] = resid[M,N] + (A@W + bias)   (f32 output)
//   RESID=false: Cb[M,N] = act(A@W + bias) cast to bf16
// Block = 256 threads = 8 waves; 128x64 C tile; each wave computes 32x32
// (2 A-frags x 2 B-frags -> 4 WMMA per K-step).
// Full tiles stage A via GLOBAL_LOAD_ASYNC_TO_LDS_B128 (ASYNCcnt); edge
// tiles fall back to clamped b128 loads + cndmask zero.  B tile is loaded
// b128 and transposed into LDS [n][k] so fragment K-pairs are contiguous.
// ACT: 0=none, 1=relu, 2=gelu(tanh)
// ---------------------------------------------------------------------------
#define BM 128
#define BN 64
#define BK 32

__device__ __forceinline__ float gelu_tanh(float x) {
    return 0.5f * x * (1.0f + tanhf(0.7978845608f * (x + 0.044715f * x * x * x)));
}

template<int ACT, bool RESID>
__global__ void gemm_bf16_kernel(const bf16_t* __restrict__ A,
                                 const bf16_t* __restrict__ W,
                                 const float*  __restrict__ bias,
                                 const float*  __restrict__ resid,
                                 bf16_t* __restrict__ Cb,
                                 float*  __restrict__ Cf,
                                 int M, int N, int K) {
    __shared__ bf16_t As[BM][BK + 8];   // [m][k]
    __shared__ bf16_t Bs[BN][BK + 8];   // [n][k]  (transposed from W)

    int n0 = blockIdx.x * BN;
    int m0 = blockIdx.y * BM;
    int tid  = threadIdx.x;
    int wave = tid >> 5;
    int lane = tid & 31;
    int wm = (wave >> 1) * 32;          // wave row offset: 0,32,64,96
    int wn = (wave & 1) * 32;           // wave col offset: 0,32
    int kb = (lane < 16) ? 0 : 8;
    int lm = lane & 15;
    bool full_tile = (m0 + BM <= M);    // uniform across block

    v8f acc00 = {}, acc01 = {}, acc10 = {}, acc11 = {};

    for (int ks = 0; ks < K; ks += BK) {
        // --- A tile: 128x32 bf16 = 512 x b128; 2 per thread ---
#if USE_ASYNC_LDS
        if (full_tile) {
#pragma unroll
            for (int j = 0; j < 2; ++j) {
                int lin = tid + 256 * j;
                int r  = lin >> 2;
                int c8 = (lin & 3) * 8;
                async_load_b128((unsigned)(size_t)&As[r][c8],
                                A + (size_t)(m0 + r) * K + ks + c8);
            }
        } else
#endif
        {
#pragma unroll
            for (int j = 0; j < 2; ++j) {
                int lin = tid + 256 * j;        // 0..511
                int r  = lin >> 2;              // 0..127
                int c8 = (lin & 3) * 8;         // 0,8,16,24
                int gr  = m0 + r;
                int grc = gr < M ? gr : M - 1;  // clamped, always-valid address
                uint4 val = *(const uint4*)(A + (size_t)grc * K + ks + c8);
                if (gr >= M) val = make_uint4(0u, 0u, 0u, 0u);
                *(uint4*)&As[r][c8] = val;
            }
        }
        // --- B tile: 32x64 bf16 = 256 x b128; 1 per thread, transpose ---
        {
            int r  = tid >> 3;              // k row 0..31
            int n8 = (tid & 7) * 8;         // n group
            uint4 val = *(const uint4*)(W + (size_t)(ks + r) * N + n0 + n8);
            union { uint4 u; bf16_t h[8]; } uv; uv.u = val;
#pragma unroll
            for (int i = 0; i < 8; ++i) Bs[n8 + i][r] = uv.h[i];
        }
        if (ks + BK < K) {
            __builtin_prefetch(A + (size_t)(m0 + (tid >> 1)) * K + ks + BK, 0, 3);
            __builtin_prefetch(W + (size_t)(ks + BK + (tid >> 3)) * N + n0, 0, 3);
        }
#if USE_ASYNC_LDS
        if (full_tile) wait_asynccnt0();
#endif
        __syncthreads();

        v16bf a0 = load_frag(&As[wm + lm][0], kb);
        v16bf a1 = load_frag(&As[wm + 16 + lm][0], kb);
        v16bf b0 = load_frag(&Bs[wn + lm][0], kb);
        v16bf b1 = load_frag(&Bs[wn + 16 + lm][0], kb);
        acc00 = WMMA_BF16(a0, b0, acc00);
        acc01 = WMMA_BF16(a0, b1, acc01);
        acc10 = WMMA_BF16(a1, b0, acc10);
        acc11 = WMMA_BF16(a1, b1, acc11);
        __syncthreads();
    }

    // Epilogue.  C layout: VGPR j -> row (j + (lane>=16 ? 8:0)); col = lane&15.
    int half8 = (lane & 16) ? 8 : 0;
#pragma unroll
    for (int mi = 0; mi < 2; ++mi) {
        v8f accA = mi ? acc10 : acc00;
        v8f accB = mi ? acc11 : acc01;
#pragma unroll
        for (int j = 0; j < 8; ++j) {
            int r = m0 + wm + mi * 16 + j + half8;
            if (r >= M) continue;
            int c0 = n0 + wn + lm;
            float v0 = accA[j] + bias[c0];
            float v1 = accB[j] + bias[c0 + 16];
            if (ACT == 1) { v0 = fmaxf(v0, 0.0f); v1 = fmaxf(v1, 0.0f); }
            if (ACT == 2) { v0 = gelu_tanh(v0);   v1 = gelu_tanh(v1);   }
            size_t o0 = (size_t)r * N + c0;
            size_t o1 = o0 + 16;
            if (RESID) {
                Cf[o0] = resid[o0] + v0;
                Cf[o1] = resid[o1] + v1;
            } else {
                Cb[o0] = (bf16_t)v0;
                Cb[o1] = (bf16_t)v1;
            }
        }
    }
}

// ---------------------------------------------------------------------------
// Flash attention (wave32 WMMA).  One workgroup = 128 query rows of one (b,h).
// 8 waves, each owns 16 query rows.  Keys processed in chunks of 32.
// Full key chunks stage K via GLOBAL_LOAD_ASYNC_TO_LDS_B128; V is loaded
// b128 and transposed.  Edge chunks (cross-attn 257 tail) use clamped loads.
// ---------------------------------------------------------------------------
template<bool CAUSAL>
__global__ void attn_kernel(const bf16_t* __restrict__ Q, int q_stride,
                            const bf16_t* __restrict__ KV, int kv_stride,
                            int k_off, int v_off,
                            bf16_t* __restrict__ Out,
                            int Tq, int nk, int kv_rows) {
    __shared__ bf16_t Ks[32][HEAD_D + 8];      // [key][hd]
    __shared__ bf16_t Vs[HEAD_D][32 + 8];      // [col][key] (transposed)
    __shared__ bf16_t Ps[8][16][32 + 8];       // per-wave P tile [row][key]

    int h = blockIdx.y, b = blockIdx.z;
    int tid  = threadIdx.x;
    int wave = tid >> 5;
    int lane = tid & 31;
    int q0 = blockIdx.x * 128;                 // tile base (within batch)
    int qw = q0 + wave * 16;                   // wave base row
    int kb = (lane < 16) ? 0 : 8;
    int lm = lane & 15;
    int half8 = (lane & 16) ? 8 : 0;
    const float scale = 0.125f;                // 1/sqrt(64)

    // Q fragments (A layout reads straight from global: K pairs contiguous)
    const bf16_t* qrow = Q + (size_t)(b * Tq + qw + lm) * q_stride + h * HEAD_D;
    v16bf aq0 = load_frag(qrow, kb);           // hd 0..31
    v16bf aq1 = load_frag(qrow + 32, kb);      // hd 32..63

    float mrow[8], lrow[8];
#pragma unroll
    for (int j = 0; j < 8; ++j) { mrow[j] = -1e30f; lrow[j] = 0.0f; }
    v8f o0 = {}, o1 = {}, o2 = {}, o3 = {};

    int kend = nk;
    if (CAUSAL) kend = (nk < q0 + 128) ? nk : (q0 + 128);

    // per-thread staging indices: 32 keys x 8 col-groups == 256 threads
    int ldr  = tid >> 3;                       // key row 0..31
    int ldc8 = (tid & 7) * 8;                  // col group 0,8,...,56

    for (int kc = 0; kc < kend; kc += 32) {
        bool full_chunk = (kc + 32 <= nk);     // uniform across block
        __syncthreads();    // previous chunk's Vs/Ks reads complete
        // ---- stage K and V chunk into LDS ----
        {
            int key  = kc + ldr;
            int keyc = key < nk ? key : nk - 1;
            size_t base = (size_t)(b * kv_rows + keyc) * kv_stride + h * HEAD_D;
#if USE_ASYNC_LDS
            if (full_chunk) {
                async_load_b128((unsigned)(size_t)&Ks[ldr][ldc8],
                                KV + base + k_off + ldc8);
            } else
#endif
            {
                uint4 k4 = *(const uint4*)(KV + base + k_off + ldc8);
                if (key >= nk) k4 = make_uint4(0u, 0u, 0u, 0u);
                *(uint4*)&Ks[ldr][ldc8] = k4;
            }
            uint4 v4 = *(const uint4*)(KV + base + v_off + ldc8);
            if (key >= nk) v4 = make_uint4(0u, 0u, 0u, 0u);
            union { uint4 u; bf16_t e[8]; } uv; uv.u = v4;
#pragma unroll
            for (int i = 0; i < 8; ++i) Vs[ldc8 + i][ldr] = uv.e[i];
        }
#if USE_ASYNC_LDS
        if (full_chunk) wait_asynccnt0();
#endif
        __syncthreads();

        // ---- S = Q K^T (16x32): two 16x16 key tiles, K-dim 64 = 2 wmma each ----
        v8f s0 = {}, s1 = {};
        {
            v16bf b00 = load_frag(&Ks[lm][0], kb);
            v16bf b01 = load_frag(&Ks[lm][32], kb);
            s0 = WMMA_BF16(aq0, b00, s0);
            s0 = WMMA_BF16(aq1, b01, s0);
            v16bf b10 = load_frag(&Ks[lm + 16][0], kb);
            v16bf b11 = load_frag(&Ks[lm + 16][32], kb);
            s1 = WMMA_BF16(aq0, b10, s1);
            s1 = WMMA_BF16(aq1, b11, s1);
        }

        // ---- online softmax update per row (rows striped over VGPR index) ----
#pragma unroll
        for (int j = 0; j < 8; ++j) {
            float v0 = s0[j] * scale;
            float v1 = s1[j] * scale;
            int key0 = kc + lm;
            if (CAUSAL) {
                int qr = qw + j + half8;
                if (key0 > qr)      v0 = -1e30f;
                if (key0 + 16 > qr) v1 = -1e30f;
            } else {
                if (key0 >= nk)      v0 = -1e30f;
                if (key0 + 16 >= nk) v1 = -1e30f;
            }
            float mx = fmaxf(v0, v1);
#pragma unroll
            for (int st = 1; st < 16; st <<= 1)
                mx = fmaxf(mx, __shfl_xor(mx, st, 32));
            float mn = fmaxf(mrow[j], mx);
            float alpha = __expf(mrow[j] - mn);
            float p0 = __expf(v0 - mn);
            float p1 = __expf(v1 - mn);
            float rs = p0 + p1;
#pragma unroll
            for (int st = 1; st < 16; st <<= 1)
                rs += __shfl_xor(rs, st, 32);
            lrow[j] = lrow[j] * alpha + rs;
            mrow[j] = mn;
            o0[j] *= alpha; o1[j] *= alpha; o2[j] *= alpha; o3[j] *= alpha;
            Ps[wave][j + half8][lm]      = (bf16_t)p0;
            Ps[wave][j + half8][lm + 16] = (bf16_t)p1;
        }
        __syncthreads();

        // ---- O += P (16x32) @ V (32x64): 4 wmma, one per 16-col tile ----
        v16bf ap  = load_frag(&Ps[wave][lm][0], kb);
        v16bf bv0 = load_frag(&Vs[lm][0], kb);
        v16bf bv1 = load_frag(&Vs[lm + 16][0], kb);
        v16bf bv2 = load_frag(&Vs[lm + 32][0], kb);
        v16bf bv3 = load_frag(&Vs[lm + 48][0], kb);
        o0 = WMMA_BF16(ap, bv0, o0);
        o1 = WMMA_BF16(ap, bv1, o1);
        o2 = WMMA_BF16(ap, bv2, o2);
        o3 = WMMA_BF16(ap, bv3, o3);
    }

    // ---- normalize and store (bf16) ----
#pragma unroll
    for (int j = 0; j < 8; ++j) {
        int r = qw + j + half8;
        float inv = 1.0f / lrow[j];
        size_t ro = (size_t)(b * Tq + r) * N_EMBD + h * HEAD_D + lm;
        Out[ro]      = (bf16_t)(o0[j] * inv);
        Out[ro + 16] = (bf16_t)(o1[j] * inv);
        Out[ro + 32] = (bf16_t)(o2[j] * inv);
        Out[ro + 48] = (bf16_t)(o3[j] * inv);
    }
}

// ---------------------------------------------------------------------------
// Host-side orchestration
// ---------------------------------------------------------------------------
extern "C" void kernel_launch(void* const* d_in, const int* in_sizes, int n_in,
                              void* d_out, int out_size, void* d_ws, size_t ws_size,
                              hipStream_t stream) {
    const float* x       = (const float*)d_in[0];
    const float* x_enc   = (const float*)d_in[1];
    const float* ln1_g   = (const float*)d_in[2];
    const float* ln1_b   = (const float*)d_in[3];
    const float* ln2_g   = (const float*)d_in[4];
    const float* ln2_b   = (const float*)d_in[5];
    const float* ln3_g   = (const float*)d_in[6];
    const float* ln3_b   = (const float*)d_in[7];
    const float* attn_w  = (const float*)d_in[8];
    const float* attn_bias = (const float*)d_in[9];
    const float* attn_proj_w = (const float*)d_in[10];
    const float* attn_proj_b = (const float*)d_in[11];
    const float* img_w   = (const float*)d_in[12];
    const float* img_b   = (const float*)d_in[13];
    const float* cap_w   = (const float*)d_in[14];
    const float* cap_b   = (const float*)d_in[15];
    const float* cross_proj_w = (const float*)d_in[16];
    const float* cross_proj_b = (const float*)d_in[17];
    const float* ad1_w   = (const float*)d_in[18];
    const float* ad1_b   = (const float*)d_in[19];
    const float* ad2_w   = (const float*)d_in[20];
    const float* ad2_b   = (const float*)d_in[21];
    const float* fc_w    = (const float*)d_in[22];
    const float* fc_b    = (const float*)d_in[23];
    const float* mproj_w = (const float*)d_in[24];
    const float* mproj_b = (const float*)d_in[25];

    const int R  = BATCH * SEQ_T;    // 8192 token rows
    const int RE = BATCH * N_ENC_;   // 2056 encoder rows
    const int D  = N_EMBD;           // 768
    const int DQ = D / 4;            // 192
    const int H4 = 4 * D;            // 3072

    char* ws = (char*)d_ws;
    size_t off = 0;
    auto alloc = [&](size_t bytes) -> void* {
        void* p = (void*)(ws + off);
        off += (bytes + 255) & ~(size_t)255;
        return p;
    };

    // bf16 weights
    bf16_t* w_attn  = (bf16_t*)alloc((size_t)D * 3 * D * 2);
    bf16_t* w_proj  = (bf16_t*)alloc((size_t)D * D * 2);
    bf16_t* w_img   = (bf16_t*)alloc((size_t)D * 2 * D * 2);
    bf16_t* w_cap   = (bf16_t*)alloc((size_t)D * D * 2);
    bf16_t* w_cross = (bf16_t*)alloc((size_t)D * D * 2);
    bf16_t* w_ad1   = (bf16_t*)alloc((size_t)D * DQ * 2);
    bf16_t* w_ad2   = (bf16_t*)alloc((size_t)DQ * D * 2);
    bf16_t* w_fc    = (bf16_t*)alloc((size_t)D * H4 * 2);
    bf16_t* w_mproj = (bf16_t*)alloc((size_t)H4 * D * 2);

    // bf16 activations
    bf16_t* hbuf  = (bf16_t*)alloc((size_t)R * D * 2);
    bf16_t* hencb = (bf16_t*)alloc((size_t)RE * D * 2);
    bf16_t* qkv   = (bf16_t*)alloc((size_t)R * 3 * D * 2);
    bf16_t* kvb   = (bf16_t*)alloc((size_t)RE * 2 * D * 2);
    bf16_t* q2    = (bf16_t*)alloc((size_t)R * D * 2);
    bf16_t* ao    = (bf16_t*)alloc((size_t)R * D * 2);
    bf16_t* po    = (bf16_t*)alloc((size_t)R * D * 2);
    bf16_t* tad   = (bf16_t*)alloc((size_t)R * DQ * 2);
    bf16_t* m1    = (bf16_t*)alloc((size_t)R * H4 * 2);
    // f32 running residual
    float*  xbuf  = (float*)alloc((size_t)R * D * 4);

    auto conv = [&](const float* src, bf16_t* dst, int n) {
        f32_to_bf16_kernel<<<(n + 255) / 256, 256, 0, stream>>>(src, dst, n);
    };
    conv(attn_w,       w_attn,  D * 3 * D);
    conv(attn_proj_w,  w_proj,  D * D);
    conv(img_w,        w_img,   D * 2 * D);
    conv(cap_w,        w_cap,   D * D);
    conv(cross_proj_w, w_cross, D * D);
    conv(ad1_w,        w_ad1,   D * DQ);
    conv(ad2_w,        w_ad2,   DQ * D);
    conv(fc_w,         w_fc,    D * H4);
    conv(mproj_w,      w_mproj, H4 * D);

    auto gemm = [&](const bf16_t* A, const bf16_t* W, const float* bias,
                    bf16_t* C, int M, int N, int K, int act) {
        dim3 g(N / BN, (M + BM - 1) / BM);
        if (act == 0)
            gemm_bf16_kernel<0, false><<<g, 256, 0, stream>>>(A, W, bias, nullptr, C, nullptr, M, N, K);
        else if (act == 1)
            gemm_bf16_kernel<1, false><<<g, 256, 0, stream>>>(A, W, bias, nullptr, C, nullptr, M, N, K);
        else
            gemm_bf16_kernel<2, false><<<g, 256, 0, stream>>>(A, W, bias, nullptr, C, nullptr, M, N, K);
    };
    auto gemm_resid = [&](const bf16_t* A, const bf16_t* W, const float* bias,
                          const float* resid, float* C, int M, int N, int K) {
        dim3 g(N / BN, (M + BM - 1) / BM);
        gemm_bf16_kernel<0, true><<<g, 256, 0, stream>>>(A, W, bias, resid, nullptr, C, M, N, K);
    };

    // ================= Stage 1: causal self-attention + adapter =============
    layernorm_kernel<<<R, 256, 0, stream>>>(x, ln1_g, ln1_b, hbuf, R);
    gemm(hbuf, w_attn, attn_bias, qkv, R, 3 * D, D, 0);
    attn_kernel<true><<<dim3(SEQ_T / 128, N_HEADS, BATCH), 256, 0, stream>>>(
        qkv, 3 * D, qkv, 3 * D, D, 2 * D, ao, SEQ_T, SEQ_T, SEQ_T);
    gemm(ao, w_proj, attn_proj_b, po, R, D, D, 0);
    gemm(po, w_ad1, ad1_b, tad, R, DQ, D, 1);                 // relu
    gemm_resid(tad, w_ad2, ad2_b, x, xbuf, R, D, DQ);         // x = x + adapter

    // ================= Stage 2: cross-attention + adapter ===================
    layernorm_kernel<<<R, 256, 0, stream>>>(xbuf, ln1_g, ln1_b, hbuf, R);
    layernorm_kernel<<<RE, 256, 0, stream>>>(x_enc, ln3_g, ln3_b, hencb, RE);
    gemm(hencb, w_img, img_b, kvb, RE, 2 * D, D, 0);
    gemm(hbuf, w_cap, cap_b, q2, R, D, D, 0);
    attn_kernel<false><<<dim3(SEQ_T / 128, N_HEADS, BATCH), 256, 0, stream>>>(
        q2, D, kvb, 2 * D, 0, D, ao, SEQ_T, N_ENC_, N_ENC_);
    gemm(ao, w_cross, cross_proj_b, po, R, D, D, 0);
    gemm(po, w_ad1, ad1_b, tad, R, DQ, D, 1);                 // relu
    gemm_resid(tad, w_ad2, ad2_b, xbuf, xbuf, R, D, DQ);      // x = x + adapter

    // ================= Stage 3: MLP + adapter ===============================
    layernorm_kernel<<<R, 256, 0, stream>>>(xbuf, ln2_g, ln2_b, hbuf, R);
    gemm(hbuf, w_fc, fc_b, m1, R, H4, D, 2);                  // gelu
    gemm(m1, w_mproj, mproj_b, po, R, D, H4, 0);
    gemm(po, w_ad1, ad1_b, tad, R, DQ, D, 1);                 // relu
    gemm_resid(tad, w_ad2, ad2_b, xbuf, (float*)d_out, R, D, DQ);

    // x_enc passes through unchanged as second tuple element
    hipMemcpyAsync((float*)d_out + (size_t)R * D, x_enc,
                   (size_t)RE * D * sizeof(float),
                   hipMemcpyDeviceToDevice, stream);
}
